// Head_selfDec_18253611008693
// MI455X (gfx1250) — matrix operand order repaired and checked by reading, not compiled, not run on hardware
//
#include <hip/hip_runtime.h>

// Problem constants (match reference): B=16, T=2048, E=512, H=64
#define B_  16
#define T_  2048
#define E_  512
#define H_  64
#define BT  (B_*T_)
#define BTH (BT*H_)

typedef __attribute__((ext_vector_type(16))) __bf16 v16bf;
typedef __attribute__((ext_vector_type(8)))  float  v8f;
typedef int v4i __attribute__((__vector_size__(16)));   // matches builtin param

// ---------------------------------------------------------------------------
// CDNA5 async global->LDS copy (ASYNCcnt-tracked), with sync fallback.
// ---------------------------------------------------------------------------
#if __has_builtin(__builtin_amdgcn_global_load_async_to_lds_b128) && \
    __has_builtin(__builtin_amdgcn_s_wait_asynccnt)
#define USE_ASYNC_LDS 1
#else
#define USE_ASYNC_LDS 0
#endif

__device__ __forceinline__ void async_copy_b128(const void* g, void* l) {
#if USE_ASYNC_LDS
    // Param 0: AS1 (global) v4i*. Param 1: LDS v4i* — build an AS3 pointer by
    // truncating the generic pointer to its low 32 bits (LDS aperture rule);
    // AS3 converts implicitly if the param is generic.
    __builtin_amdgcn_global_load_async_to_lds_b128(
        (__attribute__((address_space(1))) v4i*)(uintptr_t)g,
        (__attribute__((address_space(3))) v4i*)(uintptr_t)(uint32_t)(uintptr_t)l,
        0, 0);
#else
    *(int4*)l = *(const int4*)g;
#endif
}
__device__ __forceinline__ void wait_async_prev_block() {
#if USE_ASYNC_LDS
    __builtin_amdgcn_s_wait_asynccnt(16);   // current block done, prefetch may fly
#endif
}
__device__ __forceinline__ void wait_async_all() {
#if USE_ASYNC_LDS
    __builtin_amdgcn_s_wait_asynccnt(0);
#endif
}

// ============================================================================
// Kernel 1: QKV projection  (q|k|v)[bt,h] = x[bt,:] @ W[:,h], stored as bf16.
// grid = (BT/16/8, 3), block = 256 threads (8 waves).
// ============================================================================
__global__ __launch_bounds__(256) void qkv_proj_kernel(
    const float* __restrict__ x,
    const float* __restrict__ Wq,
    const float* __restrict__ Wk,
    const float* __restrict__ Wv,
    __bf16* __restrict__ qkv_ws)
{
    __shared__ __bf16 ldsWT[H_ * E_];   // W^T in bf16: [n][k], 64 KB

    const int sel = blockIdx.y;
    const float* W = (sel == 0) ? Wq : ((sel == 1) ? Wk : Wv);

    // Cooperative, globally-coalesced load of W (E x H f32) -> LDS as bf16 W^T
    for (int i = threadIdx.x; i < E_ * H_; i += 256) {
        int k = i >> 6;
        int n = i & 63;
        ldsWT[n * E_ + k] = (__bf16)W[i];
    }
    __syncthreads();

    const int wid  = threadIdx.x >> 5;
    const int lane = threadIdx.x & 31;
    const int h    = lane >> 4;
    const int nlo  = lane & 15;
    const int mt   = blockIdx.x * 8 + wid;

    const float* xr = x + (size_t)(mt * 16 + nlo) * E_;

    v8f acc[4];
    #pragma unroll
    for (int nt = 0; nt < 4; ++nt)
        acc[nt] = (v8f){0.f,0.f,0.f,0.f,0.f,0.f,0.f,0.f};

    for (int kc = 0; kc < E_; kc += 32) {
        v16bf a;
        #pragma unroll
        for (int j = 0; j < 8; ++j) {
            a[j]     = (__bf16)xr[kc +      8 * h + j];
            a[j + 8] = (__bf16)xr[kc + 16 + 8 * h + j];
        }
        #pragma unroll
        for (int nt = 0; nt < 4; ++nt) {
            v16bf bfrag;
            const __bf16* bp = &ldsWT[(nt * 16 + nlo) * E_ + kc + 8 * h];
            #pragma unroll
            for (int e = 0; e < 16; ++e) bfrag[e] = bp[(e & 7) + ((e >> 3) << 4)];
            acc[nt] = __builtin_amdgcn_wmma_f32_16x16x32_bf16(
                false, a, false, bfrag, (short)0, acc[nt], false, false);
        }
    }

    __bf16* dst = qkv_ws + (size_t)sel * BTH;
    #pragma unroll
    for (int nt = 0; nt < 4; ++nt) {
        #pragma unroll
        for (int r = 0; r < 8; ++r) {
            dst[(size_t)(mt * 16 + r + 8 * h) * H_ + nt * 16 + nlo] =
                (__bf16)acc[nt][r];
        }
    }
}

// ============================================================================
// Kernel 2: causal flash attention, 64-key blocks, double-buffered async V.
// One wave per 16-query tile. grid = BT/16/4 blocks, 128 threads (4 waves).
// ============================================================================
#define WPB 4
#define KB  64
__global__ __launch_bounds__(128) void attn_kernel(
    const __bf16* __restrict__ qkv_ws,
    float* __restrict__ out)
{
    __shared__ __bf16 ldsV[WPB][2][KB * H_];  // per-wave double-buffered V tile
    __shared__ __bf16 ldsP[WPB][16 * KB];     // per-wave P transpose bounce

    const int wid  = threadIdx.x >> 5;
    const int lane = threadIdx.x & 31;
    const int h    = lane >> 4;
    const int nlo  = lane & 15;

    const int gw = blockIdx.x * WPB + wid;
    const int b  = gw >> 7;
    const int qt = gw & 127;
    const int q0 = qt * 16;

    const __bf16* qw = qkv_ws;
    const __bf16* kw = qkv_ws + (size_t)BTH;
    const __bf16* vw = qkv_ws + (size_t)2 * BTH;

    // Q fragments: two A-frags covering head-dim chunks [0,32) and [32,64)
    v16bf qa[2];
    {
        const __bf16* qr = qw + (size_t)(b * T_ + q0 + nlo) * H_;
        #pragma unroll
        for (int c = 0; c < 2; ++c) {
            #pragma unroll
            for (int j = 0; j < 8; ++j) {
                qa[c][j]     = qr[c * 32 +      8 * h + j];
                qa[c][j + 8] = qr[c * 32 + 16 + 8 * h + j];
            }
        }
    }

    v8f o[4];
    #pragma unroll
    for (int nt = 0; nt < 4; ++nt)
        o[nt] = (v8f){0.f,0.f,0.f,0.f,0.f,0.f,0.f,0.f};
    float m[8], l[8];
    #pragma unroll
    for (int r = 0; r < 8; ++r) { m[r] = -3.0e38f; l[r] = 0.f; }

    const int nb = qt / 4 + 1;              // 64-key causal blocks

    // V tile stage: KB rows are contiguous in v_ws -> pure 8 KB linear copy,
    // 16 x b128 per wave (512 B per instruction across 32 lanes).
    auto stage_v = [&](int buf, int t0) {
        const __bf16* vbase = vw + (size_t)(b * T_ + t0) * H_;
        __bf16* lbase = &ldsV[wid][buf][0];
        #pragma unroll
        for (int i = 0; i < 16; ++i) {
            int off = (i * 32 + lane) * 8;  // bf16 elems; 16 B chunks
            async_copy_b128(vbase + off, lbase + off);
        }
    };

    stage_v(0, 0);                           // prefetch block 0

    for (int kb = 0; kb < nb; ++kb) {
        const int t0  = kb * KB;
        const int buf = kb & 1;
        const bool last = (kb == nb - 1);

        if (!last) stage_v(buf ^ 1, t0 + KB);   // prefetch next block

        // ---- S = Q @ K^T : 16 x 64, four 16-col subtiles -------------------
        v8f s[4];
        #pragma unroll
        for (int sub = 0; sub < 4; ++sub) {
            s[sub] = (v8f){0.f,0.f,0.f,0.f,0.f,0.f,0.f,0.f};
            const __bf16* kr = kw + (size_t)(b * T_ + t0 + sub * 16 + nlo) * H_;
            #pragma unroll
            for (int c = 0; c < 2; ++c) {
                v16bf kf;
                #pragma unroll
                for (int e = 0; e < 16; ++e) kf[e] = kr[c * 32 + 8 * h + (e & 7) + ((e >> 3) << 4)];
                s[sub] = __builtin_amdgcn_wmma_f32_16x16x32_bf16(
                    false, qa[c], false, kf, (short)0, s[sub], false, false);
            }
        }

        // ---- scale + causal mask (only last block crosses diagonal) --------
        #pragma unroll
        for (int sub = 0; sub < 4; ++sub) {
            #pragma unroll
            for (int r = 0; r < 8; ++r) {
                float v = s[sub][r] * 0.125f;   // H^-0.5
                if (last) {
                    int key  = t0 + sub * 16 + nlo;
                    int qrow = q0 + r + 8 * h;
                    if (key > qrow) v = -3.0e38f;
                }
                s[sub][r] = v;
            }
        }

        // ---- online softmax: row reductions across 16-lane groups ----------
        float alpha[8];
        #pragma unroll
        for (int r = 0; r < 8; ++r) {
            float mx = fmaxf(fmaxf(s[0][r], s[1][r]), fmaxf(s[2][r], s[3][r]));
            mx = fmaxf(mx, __shfl_xor(mx, 1, 32));
            mx = fmaxf(mx, __shfl_xor(mx, 2, 32));
            mx = fmaxf(mx, __shfl_xor(mx, 4, 32));
            mx = fmaxf(mx, __shfl_xor(mx, 8, 32));
            float nm = fmaxf(m[r], mx);
            float a  = __expf(m[r] - nm);
            float rs = 0.f;
            #pragma unroll
            for (int sub = 0; sub < 4; ++sub) {
                float p = __expf(s[sub][r] - nm);
                rs += p;
                ldsP[wid][(r + 8 * h) * KB + sub * 16 + nlo] = (__bf16)p;
            }
            rs += __shfl_xor(rs, 1, 32);
            rs += __shfl_xor(rs, 2, 32);
            rs += __shfl_xor(rs, 4, 32);
            rs += __shfl_xor(rs, 8, 32);
            l[r] = l[r] * a + rs;
            m[r] = nm;
            alpha[r] = a;
        }
        #pragma unroll
        for (int nt = 0; nt < 4; ++nt) {
            #pragma unroll
            for (int r = 0; r < 8; ++r) o[nt][r] *= alpha[r];
        }

        // ---- P A-frags from LDS (same-wave DS ops are in-order) ------------
        v16bf pa[2];
        #pragma unroll
        for (int c = 0; c < 2; ++c) {
            #pragma unroll
            for (int j = 0; j < 8; ++j) {
                pa[c][j]     = ldsP[wid][nlo * KB + c * 32 +      8 * h + j];
                pa[c][j + 8] = ldsP[wid][nlo * KB + c * 32 + 16 + 8 * h + j];
            }
        }

        // ---- wait for this block's async V copy, then O += P @ V -----------
        if (last) wait_async_all();
        else      wait_async_prev_block();

        #pragma unroll
        for (int nt = 0; nt < 4; ++nt) {
            #pragma unroll
            for (int c = 0; c < 2; ++c) {
                v16bf vb;
                #pragma unroll
                for (int e = 0; e < 16; ++e)
                    vb[e] = ldsV[wid][buf][(c * 32 + 8 * h + (e & 7) + ((e >> 3) << 4)) * H_ + nt * 16 + nlo];
                o[nt] = __builtin_amdgcn_wmma_f32_16x16x32_bf16(
                    false, pa[c], false, vb, (short)0, o[nt], false, false);
            }
        }
    }

    // ---- finalize: divide by l, store f32 ----------------------------------
    #pragma unroll
    for (int r = 0; r < 8; ++r) {
        float inv = 1.0f / l[r];
        #pragma unroll
        for (int nt = 0; nt < 4; ++nt) {
            out[(size_t)(b * T_ + q0 + r + 8 * h) * H_ + nt * 16 + nlo] =
                o[nt][r] * inv;
        }
    }
}

// ============================================================================
// Launch: K1 writes q|k|v (bf16) into d_ws (12 MB), K2 consumes them.
// ============================================================================
extern "C" void kernel_launch(void* const* d_in, const int* in_sizes, int n_in,
                              void* d_out, int out_size, void* d_ws, size_t ws_size,
                              hipStream_t stream) {
    const float* x  = (const float*)d_in[0];
    const float* Wq = (const float*)d_in[1];
    const float* Wk = (const float*)d_in[2];
    const float* Wv = (const float*)d_in[3];
    float* out = (float*)d_out;
    __bf16* qkv_ws = (__bf16*)d_ws;   // needs 3*BT*H*2 = 12 MB

    dim3 g1(BT / 16 / 8, 3);
    qkv_proj_kernel<<<g1, 256, 0, stream>>>(x, Wq, Wk, Wv, qkv_ws);

    dim3 g2(BT / 16 / WPB);
    attn_kernel<<<g2, 32 * WPB, 0, stream>>>(qkv_ws, out);
}
